// VarianceAdaptor_15272903704737
// MI455X (gfx1250) — matrix operand order repaired and checked by reading, not compiled, not run on hardware
//
#include <hip/hip_runtime.h>
#include <hip/hip_bf16.h>

// ---------------- problem constants ----------------
#define B_  16
#define S_  256
#define SP_ 258     // padded sequence length (zero row at 0 and S+1)
#define H_  512
#define T_  2048
#define FS_ 512
// K taps = 3, NB = 256 bins

typedef _Float16 v8h  __attribute__((ext_vector_type(8)));
typedef _Float16 v16h __attribute__((ext_vector_type(16)));
typedef float    v8f  __attribute__((ext_vector_type(8)));

// ---------------------------------------------------------------
// zero the two sequence-pad rows of a padded (B, SP, 512) f16 buffer
// ---------------------------------------------------------------
__global__ __launch_bounds__(256)
void zeropad_kernel(_Float16* __restrict__ buf) {
    int i = blockIdx.x * 256 + threadIdx.x;      // < B*2*512 = 16384
    int b = i >> 10;
    int j = i & 1023;
    int row = b * SP_ + ((j < 512) ? 0 : (SP_ - 1));
    buf[(size_t)row * H_ + (j & 511)] = (_Float16)0.f;
}

// ---------------------------------------------------------------
// cast f32 (B,S,512) -> f16 padded (B,SP,512)
// ---------------------------------------------------------------
__global__ __launch_bounds__(256)
void cast_f16_kernel(const float* __restrict__ in, _Float16* __restrict__ out, int n) {
    int i = blockIdx.x * 256 + threadIdx.x;
    if (i >= n) return;
    int row = i >> 9, h = i & 511;
    int b = row >> 8, s = row & 255;
    out[(size_t)(b * SP_ + s + 1) * H_ + h] = (_Float16)in[i];
}

// ---------------------------------------------------------------
// Pack conv weight w (F=512, C=512, K=3) f32 into per-lane WMMA
// B-fragment order, f16:
//   wB[k0][fTile][lane][e], k0 in [0,48): tap = k0/16, c0 = (k0%16)*32
//   lane: n = lane&15, half = lane>>4 ;  element e -> K = e + 16*half
//   value = w[f = fTile*16+n][c = c0 + e + 16*half][tap]
// Each wave then loads a whole B fragment as one contiguous 32B read.
// ---------------------------------------------------------------
__global__ __launch_bounds__(256)
void pack_wB_kernel(const float* __restrict__ w, _Float16* __restrict__ wB) {
    int o = blockIdx.x * 256 + threadIdx.x;          // < 48*32*32*16 = 786432
    if (o >= 48 * 32 * 32 * 16) return;
    int e    = o & 15;
    int lane = (o >> 4) & 31;
    int fT   = (o >> 9) & 31;
    int k0   = o >> 14;                              // 0..47
    int tap  = k0 >> 4;
    int c0   = (k0 & 15) * 32;
    int n    = lane & 15;
    int half = lane >> 4;
    int c    = c0 + e + 16 * half;
    int f    = fT * 16 + n;
    wB[o] = (_Float16)w[((size_t)f * H_ + c) * 3 + tap];
}

// ---------------------------------------------------------------
// conv1d (K=3, pad=1) as WMMA GEMM + bias + ReLU.
// Block = 64 threads = 2 waves; one 16x64 output tile per block.
// K-split: wave w accumulates k-steps [w*24, w*24+24), then wave 1
// spills its 4 v8f accumulators to LDS and wave 0 reduces + epilogue.
// Each wave: 4 accumulators (A fragment reused x4), explicit ping-pong
// double buffering so loads stay a k-step ahead of their WMMA.
// amdgpu_waves_per_eu(1,2) relaxes the register budget so the backend
// keeps both fragment buffers live instead of serializing on one
// staging set with s_wait_loadcnt 0.
// X is physically padded (B,SP,512) with zero rows -> no bounds logic,
// EXEC all-ones throughout.
// ---------------------------------------------------------------
__global__ void
__attribute__((amdgpu_flat_work_group_size(64, 64), amdgpu_waves_per_eu(1, 2)))
conv1d_wmma_kernel(const _Float16* __restrict__ X,   // (B,SP,512) f16 padded
                   const _Float16* __restrict__ WB,  // packed fragments
                   const float*    __restrict__ bias,// (512)
                   float*          __restrict__ Y)   // (B,S,512) f32
{
    __shared__ float red[1024];           // wave1: 4 accs x 8 regs x 32 lanes
    const int tid  = threadIdx.x;
    const int wv   = tid >> 5;            // 0/1 : K-split half
    const int lane = tid & 31;
    const int fT4  = blockIdx.x * 4;      // first of 4 f-tiles (n0 = fT4*16)
    const int m0   = blockIdx.y * 16;     // seq tile
    const int b    = blockIdx.z;
    const int mA   = lane & 15;
    const int half = lane >> 4;

    auto loadA = [&](int k0) -> v16h {
        const int tap = k0 >> 4;
        const int c0  = (k0 & 15) * 32;
        // padded row index: (sa + 1) = m0 + mA + tap, always in [0, SP)
        const _Float16* p =
            X + (size_t)(b * SP_ + m0 + mA + tap) * H_ + c0 + half * 16;
        v8h lo = *(const v8h*)p;
        v8h hi = *(const v8h*)(p + 8);
        return __builtin_shufflevector(lo, hi,
                 0,1,2,3,4,5,6,7,8,9,10,11,12,13,14,15);
    };
    auto loadB = [&](int k0, int j) -> v16h {
        return *(const v16h*)(WB + (((size_t)k0 * 32 + fT4 + j) * 32 + lane) * 16);
    };

    v8f acc0 = {}, acc1 = {}, acc2 = {}, acc3 = {};

    const int kBeg = wv * 24;             // 24 k-steps per wave
    const int kEnd = kBeg + 24;

    v16h aX  = loadA(kBeg);
    v16h bX0 = loadB(kBeg, 0), bX1 = loadB(kBeg, 1);
    v16h bX2 = loadB(kBeg, 2), bX3 = loadB(kBeg, 3);

    for (int k0 = kBeg; k0 < kEnd; k0 += 2) {
        v16h aY  = loadA(k0 + 1);
        v16h bY0 = loadB(k0 + 1, 0);
        v16h bY1 = loadB(k0 + 1, 1);
        v16h bY2 = loadB(k0 + 1, 2);
        v16h bY3 = loadB(k0 + 1, 3);
        acc0 = __builtin_amdgcn_wmma_f32_16x16x32_f16(false, aX, false, bX0,
                                                      (short)0, acc0, false, false);
        acc1 = __builtin_amdgcn_wmma_f32_16x16x32_f16(false, aX, false, bX1,
                                                      (short)0, acc1, false, false);
        acc2 = __builtin_amdgcn_wmma_f32_16x16x32_f16(false, aX, false, bX2,
                                                      (short)0, acc2, false, false);
        acc3 = __builtin_amdgcn_wmma_f32_16x16x32_f16(false, aX, false, bX3,
                                                      (short)0, acc3, false, false);
        const int k2 = (k0 + 2 < kEnd) ? k0 + 2 : k0 + 1;  // last: dummy reload
        aX  = loadA(k2);
        bX0 = loadB(k2, 0);
        bX1 = loadB(k2, 1);
        bX2 = loadB(k2, 2);
        bX3 = loadB(k2, 3);
        acc0 = __builtin_amdgcn_wmma_f32_16x16x32_f16(false, aY, false, bY0,
                                                      (short)0, acc0, false, false);
        acc1 = __builtin_amdgcn_wmma_f32_16x16x32_f16(false, aY, false, bY1,
                                                      (short)0, acc1, false, false);
        acc2 = __builtin_amdgcn_wmma_f32_16x16x32_f16(false, aY, false, bY2,
                                                      (short)0, acc2, false, false);
        acc3 = __builtin_amdgcn_wmma_f32_16x16x32_f16(false, aY, false, bY3,
                                                      (short)0, acc3, false, false);
    }

    // K-split reduction: wave1 -> LDS, wave0 adds + epilogue
    if (wv == 1) {
        #pragma unroll
        for (int j = 0; j < 4; ++j) {
            const v8f a = (j == 0) ? acc0 : (j == 1) ? acc1 : (j == 2) ? acc2 : acc3;
            #pragma unroll
            for (int r = 0; r < 8; ++r) red[(j * 8 + r) * 32 + lane] = a[r];
        }
    }
    __syncthreads();
    if (wv == 0) {
        // C/D layout: VGPR r -> row m0 + r + 8*half, col = lane&15
        #pragma unroll
        for (int j = 0; j < 4; ++j) {
            const v8f a = (j == 0) ? acc0 : (j == 1) ? acc1 : (j == 2) ? acc2 : acc3;
            const int f  = (fT4 + j) * 16 + mA;
            const float bv = bias[f];
            #pragma unroll
            for (int r = 0; r < 8; ++r) {
                const int s = m0 + r + half * 8;
                float v = a[r] + red[(j * 8 + r) * 32 + lane] + bv;
                v = v > 0.f ? v : 0.f;       // ReLU
                Y[(size_t)(b * S_ + s) * FS_ + f] = v;
            }
        }
    }
}

// ---------------------------------------------------------------
// LayerNorm over last dim (512), f32 (B,S,512) in -> f16 padded out
// ---------------------------------------------------------------
__global__ __launch_bounds__(256)
void layernorm_kernel(const float* __restrict__ in, const float* __restrict__ g,
                      const float* __restrict__ be, _Float16* __restrict__ out) {
    __shared__ float s1[256], s2[256];
    const int r = blockIdx.x, i = threadIdx.x;
    const float* row = in + (size_t)r * FS_;
    float a = row[i], c = row[i + 256];
    s1[i] = a + c;
    s2[i] = a * a + c * c;
    __syncthreads();
    for (int off = 128; off > 0; off >>= 1) {
        if (i < off) { s1[i] += s1[i + off]; s2[i] += s2[i + off]; }
        __syncthreads();
    }
    const float mu  = s1[0] * (1.f / 512.f);
    const float var = s2[0] * (1.f / 512.f) - mu * mu;
    const float rs  = rsqrtf(var + 1e-5f);
    const int b = r >> 8, s = r & 255;
    _Float16* orow = out + (size_t)(b * SP_ + s + 1) * H_;
    orow[i]       = (_Float16)((a - mu) * rs * g[i] + be[i]);
    orow[i + 256] = (_Float16)((c - mu) * rs * g[i + 256] + be[i + 256]);
}

// ---------------------------------------------------------------
// linear head: out[r] = dot(h[r,:512], lw) + lb   (wave32 reduction)
// h is the padded f16 buffer.
// ---------------------------------------------------------------
__global__ __launch_bounds__(128)
void linear_kernel(const _Float16* __restrict__ h, const float* __restrict__ lw,
                   const float* __restrict__ lb, float* __restrict__ out) {
    const int wv = threadIdx.x >> 5, lane = threadIdx.x & 31;
    const int r = blockIdx.x * 4 + wv;
    const int b = r >> 8, s = r & 255;
    const _Float16* row = h + (size_t)(b * SP_ + s + 1) * H_;
    float sum = 0.f;
    #pragma unroll
    for (int j = 0; j < 16; ++j) {
        int idx = j * 32 + lane;
        sum += (float)row[idx] * lw[idx];
    }
    for (int off = 16; off > 0; off >>= 1) sum += __shfl_xor(sum, off, 32);
    if (lane == 0) out[r] = sum + lb[0];     // src_mask is all-false
}

// ---------------------------------------------------------------
// x_out = x_in + emb[searchsorted(bins, tgt)] ; optional padded f16 copy
// ---------------------------------------------------------------
__global__ __launch_bounds__(256)
void add_emb_kernel(const float* __restrict__ xin, const float* __restrict__ tgt,
                    const float* __restrict__ bins, const float* __restrict__ emb,
                    float* __restrict__ xout, _Float16* __restrict__ xout16,
                    int write16) {
    __shared__ int sidx;
    const int row = blockIdx.x >> 1;
    const int hh  = (blockIdx.x & 1) * 256 + threadIdx.x;
    if (threadIdx.x == 0) {
        float t = tgt[row];
        int lo = 0, hi = 255;                 // 255 bin edges
        while (lo < hi) { int mid = (lo + hi) >> 1;
                          if (bins[mid] < t) lo = mid + 1; else hi = mid; }
        sidx = lo;
    }
    __syncthreads();
    size_t o = (size_t)row * H_ + hh;
    float v = xin[o] + emb[(size_t)sidx * H_ + hh];
    xout[o] = v;
    if (write16) {
        int b = row >> 8, s = row & 255;
        xout16[(size_t)(b * SP_ + s + 1) * H_ + hh] = (_Float16)v;
    }
}

// ---------------------------------------------------------------
// per-batch inclusive cumsum of durations (Hillis-Steele, S=256)
// ---------------------------------------------------------------
__global__ __launch_bounds__(256)
void cumsum_kernel(const int* __restrict__ dur, int* __restrict__ cum,
                   int* __restrict__ mlen, float* __restrict__ out_mlen) {
    __shared__ int sc[256];
    const int b = blockIdx.x, i = threadIdx.x;
    sc[i] = dur[b * S_ + i];
    __syncthreads();
    for (int off = 1; off < 256; off <<= 1) {
        int t = (i >= off) ? sc[i - off] : 0;
        __syncthreads();
        sc[i] += t;
        __syncthreads();
    }
    cum[b * S_ + i] = sc[i];
    if (i == 255) { mlen[b] = sc[255]; out_mlen[b] = (float)sc[255]; }
}

// ---------------------------------------------------------------
// length regulate: out[b,t,:] = valid ? x3[b, idx(b,t), :] : 0
// ---------------------------------------------------------------
__global__ __launch_bounds__(128)
void gather_kernel(const float* __restrict__ x3, const int* __restrict__ cum,
                   const int* __restrict__ mlen, float* __restrict__ out) {
    __shared__ int sidx, svalid;
    const int bt = blockIdx.x;
    const int b = bt >> 11;                   // / T_ (2048)
    const int t = bt & (T_ - 1);
    if (threadIdx.x == 0) {
        const int* c = cum + b * S_;
        int lo = 0, hi = S_;                  // count of cum <= t
        while (lo < hi) { int mid = (lo + hi) >> 1;
                          if (c[mid] <= t) lo = mid + 1; else hi = mid; }
        if (lo > S_ - 1) lo = S_ - 1;
        sidx = lo;
        svalid = (t < mlen[b]) ? 1 : 0;
    }
    __syncthreads();
    const float4* src = (const float4*)(x3 + (size_t)(b * S_ + sidx) * H_);
    float4* dst = (float4*)(out + (size_t)bt * H_);
    float4 v = make_float4(0.f, 0.f, 0.f, 0.f);
    if (svalid) v = src[threadIdx.x];
    dst[threadIdx.x] = v;
}

// ---------------------------------------------------------------
// duration -> float copy, mel_mask recompute
// ---------------------------------------------------------------
__global__ __launch_bounds__(256)
void finalize_kernel(const int* __restrict__ dur, const int* __restrict__ mlen,
                     float* __restrict__ out_dur, float* __restrict__ out_mask) {
    int i = blockIdx.x * 256 + threadIdx.x;   // < 32768
    if (i < B_ * S_) out_dur[i] = (float)dur[i];
    if (i < B_ * T_) {
        int b = i >> 11, t = i & (T_ - 1);
        out_mask[i] = (t >= mlen[b]) ? 1.f : 0.f;
    }
}

// ---------------------------------------------------------------
extern "C" void kernel_launch(void* const* d_in, const int* in_sizes, int n_in,
                              void* d_out, int out_size, void* d_ws, size_t ws_size,
                              hipStream_t stream) {
    (void)in_sizes; (void)n_in; (void)out_size; (void)ws_size;

    const float* x          = (const float*)d_in[0];
    const int*   dur_t      = (const int*)  d_in[3];
    const float* pitch_t    = (const float*)d_in[4];
    const float* energy_t   = (const float*)d_in[5];
    // params dicts flatten in insertion order: w1,b1,g1,be1,w2,b2,g2,be2,lw,lb
    const float* pitch_bins = (const float*)d_in[36];
    const float* energy_bins= (const float*)d_in[37];
    const float* pitch_emb  = (const float*)d_in[38];
    const float* energy_emb = (const float*)d_in[39];

    // workspace layout (bytes)
    char* ws = (char*)d_ws;
    _Float16* xa16   = (_Float16*)(ws + 0);           // 4.03 MB padded f16 input
    float*    bufH   = (float*)   (ws + 4227072);     // 8 MB conv scratch (later x3)
    _Float16* bufH16 = (_Float16*)(ws + 12615680);    // 4.03 MB padded LN out f16
    _Float16* wB1    = (_Float16*)(ws + 16842752);    // 1.5 MB packed weights
    _Float16* wB2    = (_Float16*)(ws + 18415616);    // 1.5 MB
    float*    x2     = (float*)   (ws + 19988480);    // 8 MB x + pitch_emb
    int*      cum    = (int*)     (ws + 28377088);    // 16 KB
    int*      mlen   = (int*)     (ws + 28393472);    // 64 B
    float*    x3     = bufH;                          // alias (bufH dead by then)

    float* out = (float*)d_out;
    const size_t OFF_PITCH  = (size_t)B_ * T_ * H_;   // 16,777,216
    const size_t OFF_ENERGY = OFF_PITCH  + B_ * S_;
    const size_t OFF_LOGDUR = OFF_ENERGY + B_ * S_;
    const size_t OFF_DUR    = OFF_LOGDUR + B_ * S_;
    const size_t OFF_MLEN   = OFF_DUR    + B_ * S_;
    const size_t OFF_MASK   = OFF_MLEN   + B_;

    auto run_pred = [&](int pb, float* pred_out) {
        const float* w1  = (const float*)d_in[pb + 0];
        const float* b1  = (const float*)d_in[pb + 1];
        const float* g1  = (const float*)d_in[pb + 2];
        const float* be1 = (const float*)d_in[pb + 3];
        const float* w2  = (const float*)d_in[pb + 4];
        const float* b2  = (const float*)d_in[pb + 5];
        const float* g2  = (const float*)d_in[pb + 6];
        const float* be2 = (const float*)d_in[pb + 7];
        const float* lw  = (const float*)d_in[pb + 8];
        const float* lb  = (const float*)d_in[pb + 9];
        pack_wB_kernel<<<3072, 256, 0, stream>>>(w1, wB1);
        pack_wB_kernel<<<3072, 256, 0, stream>>>(w2, wB2);
        conv1d_wmma_kernel<<<dim3(8, 16, 16), 64, 0, stream>>>(xa16, wB1, b1, bufH);
        layernorm_kernel<<<B_ * S_, 256, 0, stream>>>(bufH, g1, be1, bufH16);
        conv1d_wmma_kernel<<<dim3(8, 16, 16), 64, 0, stream>>>(bufH16, wB2, b2, bufH);
        layernorm_kernel<<<B_ * S_, 256, 0, stream>>>(bufH, g2, be2, bufH16);
        linear_kernel<<<(B_ * S_) / 4, 128, 0, stream>>>(bufH16, lw, lb, pred_out);
    };

    // zero the sequence-pad rows of both padded f16 buffers (ws is poisoned)
    zeropad_kernel<<<64, 256, 0, stream>>>(xa16);
    zeropad_kernel<<<64, 256, 0, stream>>>(bufH16);

    cast_f16_kernel<<<(B_ * S_ * H_) / 256, 256, 0, stream>>>(x, xa16, B_ * S_ * H_);
    run_pred(6,  out + OFF_LOGDUR);                       // duration predictor on x
    run_pred(16, out + OFF_PITCH);                        // pitch predictor on x
    add_emb_kernel<<<B_ * S_ * 2, 256, 0, stream>>>(x, pitch_t, pitch_bins,
                                                    pitch_emb, x2, xa16, 1);
    run_pred(26, out + OFF_ENERGY);                       // energy predictor on x2
    add_emb_kernel<<<B_ * S_ * 2, 256, 0, stream>>>(x2, energy_t, energy_bins,
                                                    energy_emb, x3, xa16, 0);
    cumsum_kernel<<<B_, 256, 0, stream>>>(dur_t, cum, mlen, out + OFF_MLEN);
    gather_kernel<<<B_ * T_, 128, 0, stream>>>(x3, cum, mlen, out /*OFF 0*/);
    finalize_kernel<<<(B_ * T_) / 256, 256, 0, stream>>>(dur_t, mlen,
                                                         out + OFF_DUR, out + OFF_MASK);
}
